// MultiHeadSelfAttention_13615046328816
// MI455X (gfx1250) — compile-verified, matrix operand order checked by reading, not run on hardware
//
#include <hip/hip_runtime.h>

// ---------------------------------------------------------------------------
// MHSA for MI455X (gfx1250, wave32). All heavy math via v_wmma_f32_16x16x32_bf16.
// GEMM v3: each wave owns 32 M-rows x 64 N-cols (8 WMMA per K-step); W tile
// staged into LDS with gfx1250 async-to-LDS, *double-buffered* so the fill of
// K-step i+1 overlaps the WMMAs of K-step i (one barrier per step).
// ---------------------------------------------------------------------------

typedef __attribute__((ext_vector_type(16))) __bf16 v16bf;
typedef __attribute__((ext_vector_type(8)))  __bf16 v8bf;
typedef __attribute__((ext_vector_type(4)))  __bf16 v4bf;
typedef __attribute__((ext_vector_type(8)))  float  v8f;
typedef int v4i __attribute__((vector_size(4 * sizeof(int))));

union ABf { v16bf v; v8bf h[2]; };

static constexpr int   B_  = 512;
static constexpr int   N_  = 65;
static constexpr int   C_  = 768;
static constexpr int   H_  = 12;
static constexpr int   D_  = 64;
static constexpr int   M_  = B_ * N_;        // 33280 token rows (multiple of 256)
static constexpr int   QKVC = 3 * C_;        // 2304
static constexpr float SCALE_ = 0.125f;      // 64^-0.5
static constexpr int   NP   = 80;            // 65 padded to 5x16
static constexpr int   KPAD = 96;            // softmax K padded to 3x32
static constexpr int   BTILE = 64 * 32;      // one staged W tile (elements)

#if defined(__has_builtin)
#if __has_builtin(__builtin_amdgcn_global_load_async_to_lds_b128)
#define HAVE_ASYNC_LDS 1
#endif
#endif

__device__ __forceinline__ void async_stage_b128(const __bf16* g, __bf16* l) {
#if HAVE_ASYNC_LDS
  __builtin_amdgcn_global_load_async_to_lds_b128(
      (__attribute__((address_space(1))) v4i*)g,
      (__attribute__((address_space(3))) v4i*)l, 0, 0);
#endif
}

__device__ __forceinline__ void wait_async0() {
#if HAVE_ASYNC_LDS
#if __has_builtin(__builtin_amdgcn_s_wait_asynccnt)
  __builtin_amdgcn_s_wait_asynccnt(0);
#else
  asm volatile("s_wait_asynccnt 0x0" ::: "memory");
#endif
#endif
}

// --------------------------- fp32 -> bf16 ----------------------------------
__global__ void f32_to_bf16_kern(const float* __restrict__ in,
                                 __bf16* __restrict__ out, int n) {
  int i = (blockIdx.x * blockDim.x + threadIdx.x) * 4;
  if (i + 4 <= n) {
    const float4 f = *(const float4*)(in + i);
    v4bf o;
    o[0] = (__bf16)f.x; o[1] = (__bf16)f.y; o[2] = (__bf16)f.z; o[3] = (__bf16)f.w;
    *(v4bf*)(out + i) = o;
  } else {
    for (; i < n; ++i) out[i] = (__bf16)in[i];
  }
}

// --------------------------- GEMM: y = A * W^T + bias ----------------------
// A: [M][K] bf16 row-major, W: [Ncols][K] bf16 row-major. Block = 256 thr
// (8 waves): 256 M-rows x 64 N-cols. Each wave: two 16-row tiles so each
// B fragment from LDS feeds two WMMAs. Async double-buffered W staging.
__global__ __launch_bounds__(256) void gemm_bf16_wmma(
    const __bf16* __restrict__ A,
    const __bf16* __restrict__ W,
    const float*  __restrict__ bias,
    __bf16* __restrict__ outb,   // bf16 output (or null)
    float*  __restrict__ outf,   // f32 output  (or null)
    int K, int Ncols) {
  __shared__ __align__(16) __bf16 bsh[2 * BTILE];   // ping-pong, 8 KB

  const int tid  = threadIdx.x;
  const int lane = tid & 31;
  const int wave = tid >> 5;
  const int l16  = lane & 15;
  const int hh   = lane >> 4;
  const int mBase = blockIdx.x * 256 + wave * 32;   // wave rows [mBase, mBase+32)
  const int n0    = blockIdx.y * 64;

  const int bn = tid >> 2;   // 0..63 : LDS row (output column)
  const int bc = tid & 3;    // 0..3  : 8-element K chunk
  const int stoff = bn * 32 + bc * 8;
  const __bf16* wsrc  = W + (size_t)(n0 + bn) * K + bc * 8;
  const __bf16* arow0 = A + (size_t)(mBase + l16) * K;
  const __bf16* arow1 = arow0 + (size_t)16 * K;

  v8f acc0[4] = {}, acc1[4] = {};
  const int nIt = K >> 5;

#if HAVE_ASYNC_LDS
  // prologue: fill buffer 0
  async_stage_b128(wsrc, bsh + stoff);
  wait_async0();
  __syncthreads();
#endif

  for (int it = 0; it < nIt; ++it) {
    const int k0 = it << 5;
#if HAVE_ASYNC_LDS
    const __bf16* bcur = bsh + (it & 1) * BTILE;
    if (it + 1 < nIt)   // overlap next tile's fill with this step's WMMAs
      async_stage_b128(wsrc + k0 + 32, bsh + ((it + 1) & 1) * BTILE + stoff);
#else
    const __bf16* bcur = bsh;
    *(v8bf*)(bsh + stoff) = *(const v8bf*)(wsrc + k0);
    __syncthreads();
#endif

    if (k0 + 32 < K) {
      __builtin_prefetch(arow0 + k0 + 32, 0, 3);
      __builtin_prefetch(arow1 + k0 + 32, 0, 3);
    }

    ABf a0, a1;
    a0.h[0] = *(const v8bf*)(arow0 + k0 + hh * 8);
    a0.h[1] = *(const v8bf*)(arow0 + k0 + 16 + hh * 8);
    a1.h[0] = *(const v8bf*)(arow1 + k0 + hh * 8);
    a1.h[1] = *(const v8bf*)(arow1 + k0 + 16 + hh * 8);

    const __bf16* bp = bcur + l16 * 32;
#pragma unroll
    for (int j = 0; j < 4; ++j) {
      ABf b;
      b.h[0] = *(const v8bf*)(bp + j * 512 + hh * 8);
      b.h[1] = *(const v8bf*)(bp + j * 512 + 16 + hh * 8);
      acc0[j] = __builtin_amdgcn_wmma_f32_16x16x32_bf16(
          false, a0.v, false, b.v, (short)0, acc0[j], false, false);
      acc1[j] = __builtin_amdgcn_wmma_f32_16x16x32_bf16(
          false, a1.v, false, b.v, (short)0, acc1[j], false, false);
    }

#if HAVE_ASYNC_LDS
    if (it + 1 < nIt) wait_async0();  // next buffer filled before anyone reads it
    __syncthreads();
#else
    __syncthreads();
#endif
  }

#pragma unroll
  for (int t = 0; t < 2; ++t) {
    const int mB = mBase + t * 16;
    const v8f* acc = t ? acc1 : acc0;
#pragma unroll
    for (int j = 0; j < 4; ++j) {
      const int ncol = n0 + j * 16 + l16;
      const float bv = bias[ncol];
#pragma unroll
      for (int r = 0; r < 8; ++r) {
        const int m = mB + r + hh * 8;       // C-layout: VGPR r -> row r / r+8
        const float val = acc[j][r] + bv;
        if (outf) outf[(size_t)m * Ncols + ncol] = val;
        else      outb[(size_t)m * Ncols + ncol] = (__bf16)val;
      }
    }
  }
}

// --------------------------- fused attention -------------------------------
// One block per (b, h). qkv: [M][2304] bf16 (cols 0..767=q, 768..1535=k,
// 1536..2303=v, head h at offset h*64).
__global__ __launch_bounds__(256) void attn_wmma(
    const __bf16* __restrict__ qkv,
    const float*  __restrict__ bias,     // [12][65][65]
    const float*  __restrict__ bscale,   // [1]
    __bf16* __restrict__ ao) {           // [M][768]
  __shared__ __align__(16) float  sbuf[NP * NP];     // scores (f32)
  __shared__ __align__(16) __bf16 pbuf[NP * KPAD];   // probs (bf16, K-padded)
  __shared__ __align__(16) __bf16 vt[D_ * KPAD];     // V^T [d][n]

  const int b = blockIdx.x, h = blockIdx.y;
  const int tid = threadIdx.x, lane = tid & 31, wave = tid >> 5;
  const int l16 = lane & 15, hh = lane >> 4;
  const size_t rowBase = (size_t)b * N_;

  // stage V^T (zero-padded to n=96 keys)
  for (int i = tid; i < D_ * KPAD; i += 256) {
    const int d = i / KPAD, kk = i - d * KPAD;
    __bf16 val = (__bf16)0.0f;
    if (kk < N_) val = qkv[(rowBase + kk) * QKVC + 2 * C_ + h * 64 + d];
    vt[d * KPAD + kk] = val;
  }

  // S = q @ k^T : waves 0..4 own one 16-row strip each (5x5 tiles of 16x16)
  if (wave < 5) {
    const int tr = wave;
    const int m  = tr * 16 + l16;
    ABf a0 = {}, a1 = {};
    if (m < N_) {
      const __bf16* qrow = qkv + (rowBase + m) * QKVC + h * 64;
      a0.h[0] = *(const v8bf*)(qrow + hh * 8);
      a0.h[1] = *(const v8bf*)(qrow + 16 + hh * 8);
      a1.h[0] = *(const v8bf*)(qrow + 32 + hh * 8);
      a1.h[1] = *(const v8bf*)(qrow + 48 + hh * 8);
    }
    for (int tc = 0; tc < 5; ++tc) {
      const int nc = tc * 16 + l16;
      ABf b0 = {}, b1 = {};
      if (nc < N_) {
        const __bf16* krow = qkv + (rowBase + nc) * QKVC + C_ + h * 64;
        b0.h[0] = *(const v8bf*)(krow + hh * 8);
        b0.h[1] = *(const v8bf*)(krow + 16 + hh * 8);
        b1.h[0] = *(const v8bf*)(krow + 32 + hh * 8);
        b1.h[1] = *(const v8bf*)(krow + 48 + hh * 8);
      }
      v8f acc = {};
      acc = __builtin_amdgcn_wmma_f32_16x16x32_bf16(false, a0.v, false, b0.v,
                                                    (short)0, acc, false, false);
      acc = __builtin_amdgcn_wmma_f32_16x16x32_bf16(false, a1.v, false, b1.v,
                                                    (short)0, acc, false, false);
#pragma unroll
      for (int r = 0; r < 8; ++r)
        sbuf[(tr * 16 + r + hh * 8) * NP + tc * 16 + l16] = acc[r];
    }
  }
  __syncthreads();

  // softmax(scale*S + bias_scale*bias) per row, re-quantize to bf16
  if (tid < NP) {
    const int row = tid;
    __bf16* prow = pbuf + row * KPAD;
    if (row < N_) {
      const float bs = bscale[0];
      const float* brow = bias + ((size_t)h * N_ + row) * N_;
      float* srow = sbuf + row * NP;
      float mx = -3.4e38f;
      for (int c = 0; c < N_; ++c) {
        const float s = srow[c] * SCALE_ + bs * brow[c];
        srow[c] = s;
        mx = fmaxf(mx, s);
      }
      float sum = 0.f;
      for (int c = 0; c < N_; ++c) {
        const float e = __expf(srow[c] - mx);
        srow[c] = e;
        sum += e;
      }
      const float inv = 1.f / sum;
      for (int c = 0; c < N_; ++c) prow[c] = (__bf16)(srow[c] * inv);
      for (int c = N_; c < KPAD; ++c) prow[c] = (__bf16)0.f;
    } else {
      for (int c = 0; c < KPAD; ++c) prow[c] = (__bf16)0.f;
    }
  }
  __syncthreads();

  // O = P @ V : 20 tiles (5 row-tiles x 4 col-tiles) over 8 waves
  for (int idx = wave; idx < 20; idx += 8) {
    const int tr = idx >> 2, tc = idx & 3;
    v8f acc = {};
#pragma unroll
    for (int k0 = 0; k0 < KPAD; k0 += 32) {
      ABf a, bb;
      const __bf16* prow = pbuf + (tr * 16 + l16) * KPAD + k0;
      a.h[0] = *(const v8bf*)(prow + hh * 8);
      a.h[1] = *(const v8bf*)(prow + 16 + hh * 8);
      const __bf16* vrow = vt + (tc * 16 + l16) * KPAD + k0;
      bb.h[0] = *(const v8bf*)(vrow + hh * 8);
      bb.h[1] = *(const v8bf*)(vrow + 16 + hh * 8);
      acc = __builtin_amdgcn_wmma_f32_16x16x32_bf16(false, a.v, false, bb.v,
                                                    (short)0, acc, false, false);
    }
#pragma unroll
    for (int r = 0; r < 8; ++r) {
      const int m = tr * 16 + r + hh * 8;
      if (m < N_)
        ao[(rowBase + m) * C_ + h * 64 + tc * 16 + l16] = (__bf16)acc[r];
    }
  }
}

// --------------------------- launch ----------------------------------------
extern "C" void kernel_launch(void* const* d_in, const int* in_sizes, int n_in,
                              void* d_out, int out_size, void* d_ws, size_t ws_size,
                              hipStream_t stream) {
  (void)in_sizes; (void)n_in; (void)out_size; (void)ws_size;
  const float* x      = (const float*)d_in[0];
  const float* qkv_w  = (const float*)d_in[1];
  const float* qkv_b  = (const float*)d_in[2];
  const float* proj_w = (const float*)d_in[3];
  const float* proj_b = (const float*)d_in[4];
  const float* bias   = (const float*)d_in[5];
  const float* bsc    = (const float*)d_in[6];
  float* out = (float*)d_out;

  char* ws = (char*)d_ws;
  auto alignup = [](size_t v) { return (v + 255) & ~(size_t)255; };
  size_t off = 0;
  __bf16* xb   = (__bf16*)(ws + off); off = alignup(off + (size_t)M_ * C_ * 2);
  __bf16* qwb  = (__bf16*)(ws + off); off = alignup(off + (size_t)QKVC * C_ * 2);
  __bf16* pwb  = (__bf16*)(ws + off); off = alignup(off + (size_t)C_ * C_ * 2);
  __bf16* qkvb = (__bf16*)(ws + off); off = alignup(off + (size_t)M_ * QKVC * 2);
  __bf16* aob  = (__bf16*)(ws + off); off = alignup(off + (size_t)M_ * C_ * 2);

  {
    const int n = M_ * C_;
    f32_to_bf16_kern<<<(n / 4 + 255) / 256, 256, 0, stream>>>(x, xb, n);
  }
  {
    const int n = QKVC * C_;
    f32_to_bf16_kern<<<(n / 4 + 255) / 256, 256, 0, stream>>>(qkv_w, qwb, n);
  }
  {
    const int n = C_ * C_;
    f32_to_bf16_kern<<<(n / 4 + 255) / 256, 256, 0, stream>>>(proj_w, pwb, n);
  }

  // QKV: [33280 x 768] @ [2304 x 768]^T -> bf16 [33280 x 2304]
  gemm_bf16_wmma<<<dim3(M_ / 256, QKVC / 64), 256, 0, stream>>>(
      xb, qwb, qkv_b, qkvb, nullptr, C_, QKVC);

  // fused attention per (b, h)
  attn_wmma<<<dim3(B_, H_), 256, 0, stream>>>(qkvb, bias, bsc, aob);

  // proj: [33280 x 768] @ [768 x 768]^T + b -> f32 d_out
  gemm_bf16_wmma<<<dim3(M_ / 256, C_ / 64), 256, 0, stream>>>(
      aob, pwb, proj_b, nullptr, out, C_, C_);
}
